// MPEncdecMultiheadAttn_60344290509436
// MI455X (gfx1250) — matrix operand order repaired and checked by reading, not compiled, not run on hardware
//
#include <hip/hip_runtime.h>

// ---------------- problem constants ----------------
constexpr int kE = 1024;
constexpr int kH = 16;
constexpr int kD = 64;          // kE / kH
constexpr int kR = 8;
constexpr int kT = 1024;
constexpr int kB = 8;
constexpr int kS = 1024;
constexpr int kM = kT * kB;     // 8192 token rows
constexpr float kScale = 0.125f; // D^-0.5

// ---------------- vector types ----------------
typedef __attribute__((ext_vector_type(16))) _Float16 v16h;
typedef __attribute__((ext_vector_type(8)))  _Float16 v8h;
typedef __attribute__((ext_vector_type(4)))  _Float16 v4h;
typedef __attribute__((ext_vector_type(8)))  float    v8f;

union Frag16 { v16h v; v8h h[2]; };

__device__ __forceinline__ v8f wmma_f16(v16h a, v16h b, v8f c) {
    return __builtin_amdgcn_wmma_f32_16x16x32_f16(false, a, false, b, (short)0, c,
                                                  false, false);
}

// ---------------- f32 -> f16 convert (4 elems/thread) ----------------
__global__ __launch_bounds__(256) void f32_to_f16_kernel(const float* __restrict__ in,
                                                         _Float16* __restrict__ out,
                                                         long n) {
    long i = ((long)blockIdx.x * blockDim.x + threadIdx.x) * 4;
    if (i + 3 >= n) return;
    float4 v = *(const float4*)(in + i);
    v4h o;
    o[0] = (_Float16)v.x; o[1] = (_Float16)v.y;
    o[2] = (_Float16)v.z; o[3] = (_Float16)v.w;
    *(v4h*)(out + i) = o;
}

// ---------------- rank-R weight materialization, TRANSPOSED output ----------------
// Factored weight w: (Nrows x K) rows, each elem is a rank-R dot with fac.
// Emits Wt[k*Nrows + n] = scale * dot(w[(n*K+k)*R .. +R], fac)  (f16, K x Nrows)
// 64x64 tiles, LDS transpose so both the f32 reads and f16 writes are coalesced.
__global__ __launch_bounds__(256) void make_weightsT_kernel(const float* __restrict__ w,
                                                            const float* __restrict__ fac,
                                                            _Float16* __restrict__ outT,
                                                            int Nrows, int K, float scale) {
    __shared__ _Float16 tile[64 * 65];           // [n][k] with pad
    const int n0 = blockIdx.x * 64;
    const int k0 = blockIdx.y * 64;
    const int tid = threadIdx.x;
    const int c  = tid & 63;                     // fast dim (coalesced)
    const int rq = tid >> 6;                     // 0..3

    float f[kR];
#pragma unroll
    for (int r = 0; r < kR; ++r) f[r] = fac[r];

#pragma unroll 4
    for (int rep = 0; rep < 16; ++rep) {
        const int nl = rep * 4 + rq;             // 0..63
        const float* row = w + ((long)(n0 + nl) * K + k0 + c) * kR;
        float acc = 0.f;
#pragma unroll
        for (int r = 0; r < kR; ++r) acc += row[r] * f[r];
        tile[nl * 65 + c] = (_Float16)(acc * scale);
    }
    __syncthreads();
#pragma unroll 4
    for (int rep = 0; rep < 16; ++rep) {
        const int kl = rep * 4 + rq;             // 0..63
        outT[(long)(k0 + kl) * Nrows + n0 + c] = tile[c * 65 + kl];
    }
}

// ---------------- WMMA GEMM: C[m,n] = sum_k A[m,k] * Wt[k,n] ----------------
// A: rows x K (row-major f16); Wt: K x N (row-major f16, pre-transposed weights).
// Pure global-load pipeline: no LDS, no barriers.
// Block 256 thr = 8 waves, tile 128(m) x 128(n); wave tile 32(m) x 64(n):
// per 32-K step: 2 A-frags + 4 B-frags -> 8 WMMAs.
template <bool OUT_F32>
__global__ __launch_bounds__(256) void gemm_wmma_kernel(const _Float16* __restrict__ A,
                                                        const _Float16* __restrict__ Wt,
                                                        void* __restrict__ Cout,
                                                        int K, int N) {
    const int lane = threadIdx.x & 31;
    const int wave = threadIdx.x >> 5;
    const int mbase = blockIdx.x * 128 + (wave >> 1) * 32;
    const int nbase = blockIdx.y * 128 + (wave & 1) * 64;

    v8f acc[2][4] = {};
    const int kb    = (lane < 16) ? 0 : 8;       // A-frag K sub-chunk per lane half
    const int arow0 = mbase + (lane & 15);

    for (int k0 = 0; k0 < K; k0 += 32) {
        // A fragments (lane = m row, halves = K chunks {kb..kb+7, kb+16..kb+23})
        Frag16 a[2];
#pragma unroll
        for (int ms = 0; ms < 2; ++ms) {
            const _Float16* ap = A + (long)(arow0 + ms * 16) * K + k0 + kb;
            a[ms].h[0] = *(const v8h*)ap;
            a[ms].h[1] = *(const v8h*)(ap + 16);
        }
        // B fragments straight from global: lane%16 = n col of Wt, elems = k (as A)
        const _Float16* wrow = Wt + (long)(k0 + lane) * N + nbase;
        // prefetch next K step (speculative; emits global_prefetch_b8)
        const int kpf = (k0 + 32 < K) ? 32 : 0;
        __builtin_prefetch(wrow + (long)kpf * N, 0, 0);
        __builtin_prefetch(A + (long)arow0 * K + k0 + kpf, 0, 0);
#pragma unroll
        for (int j = 0; j < 4; ++j) {
            Frag16 b;
#pragma unroll
            for (int e = 0; e < 16; ++e) {
                const int kk = (e < 8) ? (kb + e) : (kb + e + 8);
                b.v[e] = Wt[(long)(k0 + kk) * N + nbase + j * 16 + (lane & 15)];
            }
            acc[0][j] = wmma_f16(a[0].v, b.v, acc[0][j]);
            acc[1][j] = wmma_f16(a[1].v, b.v, acc[1][j]);
        }
    }

    // C layout: lane -> n = lane%16, reg r -> m = r + 8*(lane/16)
    const int ncol0 = nbase + (lane & 15);
#pragma unroll
    for (int ms = 0; ms < 2; ++ms) {
        const int mrow0 = mbase + ms * 16 + 8 * (lane >> 4);
#pragma unroll
        for (int j = 0; j < 4; ++j) {
#pragma unroll
            for (int r = 0; r < 8; ++r) {
                long idx = (long)(mrow0 + r) * N + ncol0 + j * 16;
                if (OUT_F32) ((float*)Cout)[idx] = acc[ms][j][r];
                else         ((_Float16*)Cout)[idx] = (_Float16)acc[ms][j][r];
            }
        }
    }
}

// ---------------- flash attention (f16 WMMA, S^T trick, 32 q-rows/wave) ----------------
// q16: (t*B+b) x E, stride kE.  kv16: (s*B+b) x 2E; K = cols[h*64..), V = cols[kE + h*64..).
// Computes S^T = K * Q^T so the exp'd score C-frags ARE the P A-fragment for P*V.
// Each wave: 2 t-tiles (32 q rows); K A-frags / V B-frags shared across both.
__global__ __launch_bounds__(256) void attn_kernel(const _Float16* __restrict__ q16,
                                                   const _Float16* __restrict__ kv16,
                                                   const unsigned char* __restrict__ mask,
                                                   _Float16* __restrict__ ctx16) {
    constexpr int kTT  = 2;
    constexpr int ldkv = 2 * kE;
    __shared__ _Float16 Qt[8][kTT][kD * 16];     // per-wave Q^T tiles: [d][t]

    const int lane = threadIdx.x & 31;
    const int wave = threadIdx.x >> 5;
    const int bh = blockIdx.x;                   // 0..127
    const int b  = bh / kH;
    const int h  = bh % kH;
    const int tbase = (blockIdx.y * 8 + wave) * (16 * kTT);

    // stage Q^T per t-tile: lane (t = lane&15, dpart = lane>>4) loads Q[t][dpart*32..+31]
#pragma unroll
    for (int tt = 0; tt < kTT; ++tt) {
        _Float16* qt = Qt[wave][tt];
        const int t = lane & 15, dpart = lane >> 4;
        const _Float16* qp =
            q16 + ((long)(tbase + tt * 16 + t) * kB + b) * kE + h * kD + dpart * 32;
        v8h q0 = *(const v8h*)qp;
        v8h q1 = *(const v8h*)(qp + 8);
        v8h q2 = *(const v8h*)(qp + 16);
        v8h q3 = *(const v8h*)(qp + 24);
#pragma unroll
        for (int e = 0; e < 8; ++e) {
            qt[(dpart * 32 + e)      * 16 + t] = q0[e];
            qt[(dpart * 32 + 8 + e)  * 16 + t] = q1[e];
            qt[(dpart * 32 + 16 + e) * 16 + t] = q2[e];
            qt[(dpart * 32 + 24 + e) * 16 + t] = q3[e];
        }
    }
    __syncthreads();

    // Q^T B-fragments (lane%16 = n = t, elems = k = d), per t-tile and 32-wide d chunk
    Frag16 qb[kTT][2];
#pragma unroll
    for (int tt = 0; tt < kTT; ++tt)
#pragma unroll
        for (int dd = 0; dd < 2; ++dd) {
            const _Float16* p = &Qt[wave][tt][(dd * 32) * 16];
#pragma unroll
            for (int e = 0; e < 16; ++e) {
                const int kk = (e < 8) ? ((lane >> 4) * 8 + e) : ((lane >> 4) * 8 + e + 8);
                qb[tt][dd].v[e] = p[kk * 16 + (lane & 15)];
            }
        }

    float m[kTT], l[kTT];
    v8f O[kTT][4] = {};
#pragma unroll
    for (int tt = 0; tt < kTT; ++tt) { m[tt] = -INFINITY; l[tt] = 0.f; }

    const int kb = (lane < 16) ? 0 : 8;
    const int srow_off = 8 * (lane >> 4);        // M(=s) offset of this lane's C regs

    for (int s0 = 0; s0 < kS; s0 += 32) {
        // K A-fragments, shared by both t-tiles: lane = s row, halves = d chunks
        Frag16 ka[2][2];
#pragma unroll
        for (int st = 0; st < 2; ++st) {
            const int srow = s0 + st * 16 + (lane & 15);
            const _Float16* kp = kv16 + ((long)srow * kB + b) * ldkv + h * kD;
#pragma unroll
            for (int dd = 0; dd < 2; ++dd) {
                ka[st][dd].h[0] = *(const v8h*)(kp + dd * 32 + kb);
                ka[st][dd].h[1] = *(const v8h*)(kp + dd * 32 + kb + 16);
            }
        }
        // mask bytes for this lane's 16 s values (shared by both t-tiles)
        unsigned char mk[2][8];
#pragma unroll
        for (int st = 0; st < 2; ++st) {
            const unsigned char* mp = mask + b * kS + s0 + st * 16 + srow_off;
#pragma unroll
            for (int r = 0; r < 8; ++r) mk[st][r] = mp[r];
        }

        Frag16 pa[kTT];
#pragma unroll
        for (int tt = 0; tt < kTT; ++tt) {
            // S^T tiles: sc[st] = s rows (s0+st*16..+15) x t cols (0..15)
            v8f sc[2] = {};
#pragma unroll
            for (int st = 0; st < 2; ++st)
#pragma unroll
                for (int dd = 0; dd < 2; ++dd)
                    sc[st] = wmma_f16(ka[st][dd].v, qb[tt][dd].v, sc[st]);

            float tmax = -INFINITY;
#pragma unroll
            for (int st = 0; st < 2; ++st)
#pragma unroll
                for (int r = 0; r < 8; ++r) {
                    float v = sc[st][r];
                    if (mk[st][r]) v = -3.0e38f;
                    sc[st][r] = v;
                    tmax = fmaxf(tmax, v);
                }
            tmax = fmaxf(tmax, __shfl_xor(tmax, 16, 32));
            const float mnew   = fmaxf(m[tt], tmax);
            const float factor = __expf(m[tt] - mnew);
            m[tt] = mnew;

            float rsum = 0.f;
#pragma unroll
            for (int st = 0; st < 2; ++st)
#pragma unroll
                for (int r = 0; r < 8; ++r) {
                    float p = __expf(sc[st][r] - mnew);
                    rsum += p;
                    pa[tt].h[st][r] = (_Float16)p;   // == P A-frag layout
                }
            rsum += __shfl_xor(rsum, 16, 32);
            l[tt] = l[tt] * factor + rsum;

            // rescale O: reg r belongs to row t = r + srow_off; factor lives in lane t
#pragma unroll
            for (int r = 0; r < 8; ++r) {
                float fr = __shfl(factor, r + srow_off, 32);
#pragma unroll
                for (int j = 0; j < 4; ++j) O[tt][j][r] *= fr;
            }
        }

        // O += P * V : V B-frag (lane%16 = d, elems = s) shared by both t-tiles
#pragma unroll
        for (int j = 0; j < 4; ++j) {
            Frag16 vb;
#pragma unroll
            for (int e = 0; e < 16; ++e) {
                const int kk = (e < 8) ? (kb + e) : (kb + e + 8);
                vb.v[e] = kv16[((long)(s0 + kk) * kB + b) * ldkv + kE + h * kD + j * 16 + (lane & 15)];
            }
#pragma unroll
            for (int tt = 0; tt < kTT; ++tt)
                O[tt][j] = wmma_f16(pa[tt].v, vb.v, O[tt][j]);
        }
    }

    // finalize: divide by l, store ctx16[t, b, h*64 + d]
#pragma unroll
    for (int tt = 0; tt < kTT; ++tt)
#pragma unroll
        for (int r = 0; r < 8; ++r) {
            float inv = 1.0f / __shfl(l[tt], r + srow_off, 32);
            const int t = tbase + tt * 16 + r + srow_off;
#pragma unroll
            for (int j = 0; j < 4; ++j) {
                const int d = h * kD + j * 16 + (lane & 15);
                ctx16[((long)t * kB + b) * kE + d] = (_Float16)(O[tt][j][r] * inv);
            }
        }
}

// ---------------- launcher ----------------
extern "C" void kernel_launch(void* const* d_in, const int* in_sizes, int n_in,
                              void* d_out, int out_size, void* d_ws, size_t ws_size,
                              hipStream_t stream) {
    const float* query = (const float*)d_in[0];          // (T,B,E)
    const float* key   = (const float*)d_in[1];          // (S,B,E)
    const float* tgt   = (const float*)d_in[2];          // (R,)
    const float* src   = (const float*)d_in[3];          // (R,)
    const float* wqf   = (const float*)d_in[4];          // (E*E,R)
    const float* wkvf  = (const float*)d_in[5];          // (2E*E,R)
    const float* wof   = (const float*)d_in[6];          // (E*E,R)
    const unsigned char* mask = (const unsigned char*)d_in[7]; // (B,S) bool
    float* out = (float*)d_out;

    const long MM = (long)kM * kE;        // 8M elems per activation matrix
    const long WW = (long)kE * kE;        // 1M elems per weight matrix

    _Float16* ws      = (_Float16*)d_ws;
    _Float16* query16 = ws;
    _Float16* key16   = ws + MM;
    _Float16* q16     = ws + 2 * MM;
    _Float16* kv16    = ws + 3 * MM;      // 8192 x 2048 (K | V)
    _Float16* ctx16   = ws + 5 * MM;
    _Float16* wqT     = ws + 6 * MM;      // 1024 x 1024, pre-transposed (K x N)
    _Float16* wkvT    = wqT + WW;         // 1024 x 2048
    _Float16* woT     = wkvT + 2 * WW;    // 1024 x 1024
    if (ws_size < (size_t)(6 * MM + 4 * WW) * sizeof(_Float16)) return;

    // 1) f32 -> f16 conversions of the activations
    f32_to_f16_kernel<<<(int)(MM / 1024), 256, 0, stream>>>(query, query16, MM);
    f32_to_f16_kernel<<<(int)(MM / 1024), 256, 0, stream>>>(key, key16, MM);

    // 2) materialize rank-8 weights, transposed to K x N (SCALE folded into wq)
    make_weightsT_kernel<<<dim3(kE / 64, kE / 64), 256, 0, stream>>>(wqf, tgt, wqT,
                                                                     kE, kE, kScale);
    make_weightsT_kernel<<<dim3(2 * kE / 64, kE / 64), 256, 0, stream>>>(wkvf, src, wkvT,
                                                                         2 * kE, kE, 1.0f);
    make_weightsT_kernel<<<dim3(kE / 64, kE / 64), 256, 0, stream>>>(wof, tgt, woT,
                                                                     kE, kE, 1.0f);

    // 3) projections (pure-WMMA GEMMs, K/V fused into one N=2048 pass)
    gemm_wmma_kernel<false><<<dim3(kM / 128, kE / 128), 256, 0, stream>>>(
        query16, wqT, q16, kE, kE);
    gemm_wmma_kernel<false><<<dim3(kM / 128, 2 * kE / 128), 256, 0, stream>>>(
        key16, wkvT, kv16, kE, 2 * kE);

    // 4) flash attention: grid = (B*H, T / (32 rows * 8 waves))
    attn_kernel<<<dim3(kB * kH, kT / 256), 256, 0, stream>>>(q16, kv16, mask, ctx16);

    // 5) output projection, f32 result straight into d_out
    gemm_wmma_kernel<true><<<dim3(kM / 128, kE / 128), 256, 0, stream>>>(
        ctx16, woT, out, kE, kE);
}
